// EvidentialMLPEdge_26603027432206
// MI455X (gfx1250) — matrix operand order repaired and checked by reading, not compiled, not run on hardware
//
#include <hip/hip_runtime.h>
#include <hip/hip_bf16.h>
#include <math.h>

#define N_NODES 50000
#define N_EDGES 800000
#define IN_DIM  256
#define HID     64
#define ZPAD    68          // LDS row pad (floats): conflict-free, 16B-aligned
#define WSW_ELEMS (8*4*32*16)  // swizzled weight fragments: [s][t][lane][e]

typedef __attribute__((ext_vector_type(16))) __bf16 v16bf;
typedef __attribute__((ext_vector_type(8)))  float  v8f;

__device__ __forceinline__ float softplus_f(float x) {
    return (x > 30.f) ? x : log1pf(expf(x));
}

// |x| via opaque inline asm: the gfx1250 backend crashes selecting the
// vectorized llvm.fabs sign-mask pattern, and every branch-free C formulation
// (fabsf, maxnum(x,-x), int and-mask) is canonicalized back to llvm.fabs.
// A raw v_and_b32 cannot be recognized or vectorized.
__device__ __forceinline__ float abs_safe(float x) {
    float r;
    asm("v_and_b32 %0, 0x7fffffff, %1" : "=v"(r) : "v"(x));
    return r;
}

// Pre-swizzle a [256][64] fp32 weight matrix into bf16 B-fragment layout:
// frag (s,t): lane L element e holds W[32*s + (L/16)*8 + (e%8) + (e/8)*16][16*t + (L%16)]
__global__ __launch_bounds__(256) void swizzle_kernel(
    const float* __restrict__ w1, const float* __restrict__ pw,
    __bf16* __restrict__ w1sw, __bf16* __restrict__ pwsw)
{
    int idx = blockIdx.x * 256 + threadIdx.x;
    if (idx >= WSW_ELEMS) return;
    int e = idx & 15;
    int L = (idx >> 4) & 31;
    int t = (idx >> 9) & 3;
    int s = idx >> 11;
    int k = 32 * s + (L >> 4) * 8 + (e & 7) + (e >> 3) * 16;
    int n = 16 * t + (L & 15);
    w1sw[idx] = (__bf16)w1[k * HID + n];
    pwsw[idx] = (__bf16)pw[k * HID + n];
}

// h = tanh(node_x @ proj_w + proj_b); one wave per 16-node tile.
// 3125 tiles = 625 blocks * 5 waves.
__global__ __launch_bounds__(160) void proj_kernel(
    const float* __restrict__ x, const __bf16* __restrict__ wsw,
    const float* __restrict__ b, float* __restrict__ h)
{
    const int lane = threadIdx.x & 31;
    const int wave = threadIdx.x >> 5;
    const int tile = blockIdx.x * 5 + wave;
    const int m    = lane & 15;
    const int half = lane >> 4;
    const float* xr = x + (size_t)(tile * 16 + m) * IN_DIM;

    v8f acc[4] = {v8f{}, v8f{}, v8f{}, v8f{}};

    #pragma unroll
    for (int s = 0; s < 8; ++s) {
        const int k0 = 32 * s + half * 8;          // elements 0..7
        float4 a0 = *(const float4*)(xr + k0);
        float4 a1 = *(const float4*)(xr + k0 + 4);
        float4 c0 = *(const float4*)(xr + k0 + 16); // elements 8..15
        float4 c1 = *(const float4*)(xr + k0 + 20);
        v16bf A;
        A[0]  = (__bf16)a0.x; A[1]  = (__bf16)a0.y; A[2]  = (__bf16)a0.z; A[3]  = (__bf16)a0.w;
        A[4]  = (__bf16)a1.x; A[5]  = (__bf16)a1.y; A[6]  = (__bf16)a1.z; A[7]  = (__bf16)a1.w;
        A[8]  = (__bf16)c0.x; A[9]  = (__bf16)c0.y; A[10] = (__bf16)c0.z; A[11] = (__bf16)c0.w;
        A[12] = (__bf16)c1.x; A[13] = (__bf16)c1.y; A[14] = (__bf16)c1.z; A[15] = (__bf16)c1.w;
        #pragma unroll
        for (int t = 0; t < 4; ++t) {
            v16bf B = *(const v16bf*)(wsw + (size_t)((s * 4 + t) * 32 + lane) * 16);
            acc[t] = __builtin_amdgcn_wmma_f32_16x16x32_bf16(
                false, A, false, B, (short)0, acc[t], false, false);
        }
    }

    #pragma unroll
    for (int t = 0; t < 4; ++t) {
        const int n = 16 * t + m;
        const float bn = b[n];
        #pragma unroll
        for (int r = 0; r < 8; ++r) {
            const int mm = half ? (r + 8) : r;
            h[(size_t)(tile * 16 + mm) * HID + n] = tanhf(acc[t][r] + bn);
        }
    }
}

// Edge MLP + evidential head; one wave per 16-edge tile.
// 50000 tiles = 12500 blocks * 4 waves.
__global__ __launch_bounds__(128) void edge_kernel(
    const long long* __restrict__ eidx, const float* __restrict__ h,
    const __bf16* __restrict__ wsw, const float* __restrict__ b1,
    const float* __restrict__ w2, const float* __restrict__ b2,
    float* __restrict__ out)
{
    __shared__ float shZ[4][2][16][ZPAD];   // per-wave z_u / z_v staging, ~34 KB

    const int lane  = threadIdx.x & 31;
    const int wave  = threadIdx.x >> 5;
    const int tile  = blockIdx.x * 4 + wave;
    const int ebase = tile * 16;
    const int m     = lane & 15;
    const int half  = lane >> 4;

    // lanes 0-15 hold u-index of edge m, lanes 16-31 hold v-index (fits 32-bit)
    const int myNode = (int)eidx[(size_t)half * N_EDGES + ebase + m];

    // cooperative gather: 16 edges * 2 nodes * 64 floats -> LDS (float4 granularity)
    #pragma unroll
    for (int i = 0; i < 16; ++i) {
        int flat  = lane + 32 * i;       // 0..511
        int edge  = flat >> 5;
        int which = (flat >> 4) & 1;
        int c4    = flat & 15;
        int node  = __shfl(myNode, which * 16 + edge, 32);
        float4 v  = *(const float4*)(h + (size_t)node * HID + c4 * 4);
        *(float4*)&shZ[wave][which][edge][c4 * 4] = v;
    }
    __syncthreads();

    const float* zu = &shZ[wave][0][m][0];
    const float* zv = &shZ[wave][1][m][0];

    v8f acc[4] = {v8f{}, v8f{}, v8f{}, v8f{}};

    #pragma unroll
    for (int s = 0; s < 8; ++s) {
        const int sec = s >> 1;                 // 0:z_u 1:z_v 2:|z_u-z_v| 3:z_u*z_v
        const int k0  = 32 * s + half * 8;
        const int c0  = k0 & 63;
        const int c1  = (k0 + 16) & 63;
        float f[16];
        {   // chunk 0 -> elements 0..7
            float4 u0 = *(const float4*)(zu + c0), u1 = *(const float4*)(zu + c0 + 4);
            float4 v0 = *(const float4*)(zv + c0), v1 = *(const float4*)(zv + c0 + 4);
            float uu[8] = {u0.x,u0.y,u0.z,u0.w,u1.x,u1.y,u1.z,u1.w};
            float vv[8] = {v0.x,v0.y,v0.z,v0.w,v1.x,v1.y,v1.z,v1.w};
            #pragma unroll
            for (int i = 0; i < 8; ++i)
                f[i] = (sec == 0) ? uu[i] : (sec == 1) ? vv[i]
                     : (sec == 2) ? abs_safe(uu[i] - vv[i]) : uu[i] * vv[i];
        }
        {   // chunk 1 -> elements 8..15
            float4 u0 = *(const float4*)(zu + c1), u1 = *(const float4*)(zu + c1 + 4);
            float4 v0 = *(const float4*)(zv + c1), v1 = *(const float4*)(zv + c1 + 4);
            float uu[8] = {u0.x,u0.y,u0.z,u0.w,u1.x,u1.y,u1.z,u1.w};
            float vv[8] = {v0.x,v0.y,v0.z,v0.w,v1.x,v1.y,v1.z,v1.w};
            #pragma unroll
            for (int i = 0; i < 8; ++i)
                f[8 + i] = (sec == 0) ? uu[i] : (sec == 1) ? vv[i]
                         : (sec == 2) ? abs_safe(uu[i] - vv[i]) : uu[i] * vv[i];
        }
        v16bf A;
        #pragma unroll
        for (int i = 0; i < 16; ++i) A[i] = (__bf16)f[i];

        #pragma unroll
        for (int t = 0; t < 4; ++t) {
            v16bf B = *(const v16bf*)(wsw + (size_t)((s * 4 + t) * 32 + lane) * 16);
            acc[t] = __builtin_amdgcn_wmma_f32_16x16x32_bf16(
                false, A, false, B, (short)0, acc[t], false, false);
        }
    }

    // bias + ReLU, stage hidden tile (16x64) through LDS (reuse wave's region)
    float* shH = &shZ[wave][0][0][0];
    #pragma unroll
    for (int t = 0; t < 4; ++t) {
        const int n = 16 * t + m;
        const float bn = b1[n];
        #pragma unroll
        for (int r = 0; r < 8; ++r) {
            const int mm = half ? (r + 8) : r;
            shH[mm * ZPAD + n] = fmaxf(acc[t][r] + bn, 0.f);
        }
    }
    __syncthreads();

    // 64 -> 2 head + evidential outputs: lanes 0-15, one edge each
    if (lane < 16) {
        const float* hr = shH + lane * ZPAD;
        float l0 = b2[0], l1 = b2[1];
        #pragma unroll 8
        for (int n = 0; n < HID; ++n) {
            float hv = hr[n];
            float2 wn = *(const float2*)(w2 + 2 * n);
            l0 = fmaf(hv, wn.x, l0);
            l1 = fmaf(hv, wn.y, l1);
        }
        float e0 = softplus_f(l0), e1 = softplus_f(l1);
        float a0 = e0 + 1.f, a1 = e1 + 1.f;
        float S  = a0 + a1;
        size_t eg = (size_t)ebase + lane;
        out[2 * eg]     = e0;                         // evidence
        out[2 * eg + 1] = e1;
        out[(size_t)2 * N_EDGES + 2 * eg]     = a0;   // alpha
        out[(size_t)2 * N_EDGES + 2 * eg + 1] = a1;
        out[(size_t)4 * N_EDGES + eg] = 2.f / (S + 1e-12f);  // uncertainty
        out[(size_t)5 * N_EDGES + 2 * eg]     = a0 / S;      // probs
        out[(size_t)5 * N_EDGES + 2 * eg + 1] = a1 / S;
    }
}

extern "C" void kernel_launch(void* const* d_in, const int* in_sizes, int n_in,
                              void* d_out, int out_size, void* d_ws, size_t ws_size,
                              hipStream_t stream) {
    const long long* eidx  = (const long long*)d_in[0];   // int64 [2, E]
    const float* node_x    = (const float*)d_in[1];       // [N, 256]
    const float* proj_w    = (const float*)d_in[2];       // [256, 64]
    const float* proj_b    = (const float*)d_in[3];       // [64]
    const float* w1        = (const float*)d_in[4];       // [256, 64]
    const float* b1        = (const float*)d_in[5];       // [64]
    const float* w2        = (const float*)d_in[6];       // [64, 2]
    const float* b2        = (const float*)d_in[7];       // [2]
    float* out = (float*)d_out;

    // workspace: h [N*64 f32] | w1 swizzled bf16 | proj_w swizzled bf16
    float*  h     = (float*)d_ws;
    __bf16* w1sw  = (__bf16*)((char*)d_ws + (size_t)N_NODES * HID * sizeof(float));
    __bf16* pwsw  = w1sw + WSW_ELEMS;

    swizzle_kernel<<<(WSW_ELEMS + 255) / 256, 256, 0, stream>>>(w1, proj_w, w1sw, pwsw);
    proj_kernel<<<(N_NODES / 16) / 5, 160, 0, stream>>>(node_x, pwsw, proj_b, h);
    edge_kernel<<<(N_EDGES / 16) / 4, 128, 0, stream>>>(eidx, h, w1sw, b1, w2, b2, out);
}
